// LocalStridedBlockSparseAttnInferenceBT_687194767504
// MI455X (gfx1250) — matrix operand order repaired and checked by reading, not compiled, not run on hardware
//
#include <hip/hip_runtime.h>

typedef __attribute__((ext_vector_type(16))) _Float16 v16h;
typedef __attribute__((ext_vector_type(8)))  float    v8f;

#define B_    16
#define H_    16
#define KVH_  4
#define D_    128
#define BLK_  16
#define MAXB_ 256
#define NWAVE 4

// Load a 16x32 f16 A-tile slice for the score WMMA.
// Lane (half hs, row tok) needs dims {dbase..dbase+7} in a[0..7] and
// {dbase+16..dbase+23} in a[8..15]  (dbase = chunk*32 + hs*8).
// k_cache block layout: float offset (d>>2)*64 + tok*4 + (d&3)  -> float4: (d>>2)*16 + tok
__device__ __forceinline__ v16h load_k_tile(const float4* __restrict__ K4, int dbase, int tok) {
  float4 f0 = K4[((dbase     ) >> 2) * 16 + tok];
  float4 f1 = K4[((dbase + 4 ) >> 2) * 16 + tok];
  float4 f2 = K4[((dbase + 16) >> 2) * 16 + tok];
  float4 f3 = K4[((dbase + 20) >> 2) * 16 + tok];
  v16h a;
  a[0]=(_Float16)f0.x; a[1]=(_Float16)f0.y; a[2]=(_Float16)f0.z; a[3]=(_Float16)f0.w;
  a[4]=(_Float16)f1.x; a[5]=(_Float16)f1.y; a[6]=(_Float16)f1.z; a[7]=(_Float16)f1.w;
  a[8]=(_Float16)f2.x; a[9]=(_Float16)f2.y; a[10]=(_Float16)f2.z; a[11]=(_Float16)f2.w;
  a[12]=(_Float16)f3.x; a[13]=(_Float16)f3.y; a[14]=(_Float16)f3.z; a[15]=(_Float16)f3.w;
  return a;
}

__global__ __launch_bounds__(128, 1) void sparse_attn_decode(
    const float* __restrict__ qg,
    const float* __restrict__ k_cache,
    const float* __restrict__ v_cache,
    const int*   __restrict__ block_tables,
    const int*   __restrict__ context_lens,
    const int*   __restrict__ crow,
    const int*   __restrict__ colv,
    int col_stride,
    float* __restrict__ outp)
{
  const int h    = blockIdx.x;
  const int b    = blockIdx.y;
  const int tid  = threadIdx.x;
  const int wave = tid >> 5;
  const int lane = tid & 31;
  const int hs   = lane >> 4;   // wave half select (0/1)
  const int tok  = lane & 15;   // matrix M row for this lane

  __shared__ _Float16 sh_q[D_];
  __shared__ float    sh_O[NWAVE][D_];
  __shared__ float    sh_ms[NWAVE][2];

  const float sm_scale = 0.088388347648318447f; // 1/sqrt(128)
  sh_q[tid] = (_Float16)(qg[(b * H_ + h) * D_ + tid] * sm_scale);
  __syncthreads();

  const int qpid  = context_lens[b] - 1;
  const int pbid  = qpid >> 4;
  const int start = crow[h * (MAXB_ + 1) + pbid];
  const int end   = crow[h * (MAXB_ + 1) + pbid + 1];
  const int kvh   = h >> 2;   // h / (H/KVH)

  // B operand for scores: B[k,n] = qh[chunk*32 + 8*hs + k + (k & 8)] for all n.
  v16h bq[4];
  #pragma unroll
  for (int c = 0; c < 4; ++c) {
    const int base = c * 32 + hs * 8;
    #pragma unroll
    for (int k = 0; k < 16; ++k) bq[c][k] = sh_q[base + k + (k & 8)];
  }

  const float NEGINF = -__builtin_inff();
  float m_run = NEGINF;
  float s_run = 0.f;
  v8f zero8 = {};
  v8f oacc[8];
  #pragma unroll
  for (int i = 0; i < 8; ++i) oacc[i] = zero8;

  // Each wave handles block pairs p = wave, wave+4, ... (flash-decoding split).
  for (int p = wave; start + 2 * p < end; p += NWAVE) {
    const int  j0   = start + 2 * p;
    const bool has1 = (j0 + 1) < end;
    const int  c0   = colv[h * col_stride + j0];
    const int  c1   = has1 ? colv[h * col_stride + j0 + 1] : c0;
    const long bt0  = block_tables[b * MAXB_ + c0];
    const long bt1  = block_tables[b * MAXB_ + c1];
    const float4* K0 = (const float4*)(k_cache + (bt0 * KVH_ + kvh) * 2048);
    const float4* K1 = (const float4*)(k_cache + (bt1 * KVH_ + kvh) * 2048);
    const float4* V0 = (const float4*)(v_cache + (bt0 * KVH_ + kvh) * 2048);
    const float4* V1 = (const float4*)(v_cache + (bt1 * KVH_ + kvh) * 2048);

    // Prefetch next pair's first block (8KB tile = lane-spread 256B strides).
    {
      const int jn = j0 + 2 * NWAVE;
      if (jn < end) {
        const int  cn  = colv[h * col_stride + jn];
        const long btn = block_tables[b * MAXB_ + cn];
        const float* Kn = k_cache + (btn * KVH_ + kvh) * 2048;
        const float* Vn = v_cache + (btn * KVH_ + kvh) * 2048;
        __builtin_prefetch(Kn + lane * 64, 0, 3);
        __builtin_prefetch(Vn + lane * 64, 0, 3);
      }
    }

    // ---- scores: S[t] = sum_d qh[d]*K[d][t], via 4 K-chunk WMMAs per block ----
    v8f s0 = zero8, s1 = zero8;
    #pragma unroll
    for (int c = 0; c < 4; ++c) {
      const int dbase = c * 32 + hs * 8;
      v16h a0 = load_k_tile(K0, dbase, tok);
      v16h a1 = load_k_tile(K1, dbase, tok);
      s0 = __builtin_amdgcn_wmma_f32_16x16x32_f16(false, a0, false, bq[c], (short)0, s0, false, false);
      s1 = __builtin_amdgcn_wmma_f32_16x16x32_f16(false, a1, false, bq[c], (short)0, s1, false, false);
    }
    // lane half hs holds s[token r + 8*hs] in reg r (all 16 D-columns identical)

    // ---- masked online softmax over the 32 scores of this pair ----
    float p0[8], p1[8];
    float mloc = NEGINF;
    #pragma unroll
    for (int r = 0; r < 8; ++r) {
      const int t  = r + 8 * hs;
      float x0 = (c0 * BLK_ + t <= qpid)          ? s0[r] : NEGINF;
      float x1 = (has1 && (c1 * BLK_ + t <= qpid)) ? s1[r] : NEGINF;
      p0[r] = x0; p1[r] = x1;
      mloc  = fmaxf(mloc, fmaxf(x0, x1));
    }
    mloc = fmaxf(mloc, __shfl_xor(mloc, 16, 32));
    const float m_new = fmaxf(m_run, mloc);
    const bool  dead  = (m_new == NEGINF);
    const float alpha = dead ? 1.f : __expf(m_run - m_new);
    float lsum = 0.f;
    #pragma unroll
    for (int r = 0; r < 8; ++r) {
      p0[r] = (dead || p0[r] == NEGINF) ? 0.f : __expf(p0[r] - m_new);
      p1[r] = (dead || p1[r] == NEGINF) ? 0.f : __expf(p1[r] - m_new);
      lsum += p0[r] + p1[r];
    }
    lsum  += __shfl_xor(lsum, 16, 32);
    s_run  = s_run * alpha + lsum;
    m_run  = m_new;

    // rescale running output accumulators
    #pragma unroll
    for (int dc = 0; dc < 8; ++dc)
      #pragma unroll
      for (int r = 0; r < 8; ++r) oacc[dc][r] *= alpha;

    // ---- build PV B operand: each half carries its own p0[0..7] in [0..7] and p1[0..7] in [8..15] ----
    v16h pb;
    #pragma unroll
    for (int r = 0; r < 8; ++r) {
      pb[r]     = (_Float16)p0[r];
      pb[r + 8] = (_Float16)p1[r];
    }

    // ---- O[d] += sum_t p[t]*V[d][t], K=32 covers both blocks, 8 d-chunks ----
    #pragma unroll
    for (int dc = 0; dc < 8; ++dc) {
      const int d = dc * 16 + tok;
      float4 a0 = V0[d * 4 + 2 * hs];
      float4 a1 = V0[d * 4 + 2 * hs + 1];
      float4 b0 = V1[d * 4 + 2 * hs];
      float4 b1 = V1[d * 4 + 2 * hs + 1];
      v16h av;
      av[0]=(_Float16)a0.x; av[1]=(_Float16)a0.y; av[2]=(_Float16)a0.z; av[3]=(_Float16)a0.w;
      av[4]=(_Float16)a1.x; av[5]=(_Float16)a1.y; av[6]=(_Float16)a1.z; av[7]=(_Float16)a1.w;
      av[8]=(_Float16)b0.x; av[9]=(_Float16)b0.y; av[10]=(_Float16)b0.z; av[11]=(_Float16)b0.w;
      av[12]=(_Float16)b1.x; av[13]=(_Float16)b1.y; av[14]=(_Float16)b1.z; av[15]=(_Float16)b1.w;
      oacc[dc] = __builtin_amdgcn_wmma_f32_16x16x32_f16(false, av, false, pb, (short)0, oacc[dc], false, false);
    }
  }

  // ---- per-wave partials -> LDS (lane 0 holds O[16dc+r], lane 16 holds O[16dc+8+r]) ----
  if (lane == 0 || lane == 16) {
    #pragma unroll
    for (int dc = 0; dc < 8; ++dc)
      #pragma unroll
      for (int r = 0; r < 8; ++r)
        sh_O[wave][dc * 16 + 8 * hs + r] = oacc[dc][r];
  }
  if (lane == 0) { sh_ms[wave][0] = m_run; sh_ms[wave][1] = s_run; }
  __syncthreads();

  // ---- combine the 4 wave partials (one thread per output element d = tid) ----
  float M = NEGINF;
  #pragma unroll
  for (int w = 0; w < NWAVE; ++w) M = fmaxf(M, sh_ms[w][0]);
  float S = 0.f, O = 0.f;
  #pragma unroll
  for (int w = 0; w < NWAVE; ++w) {
    const float mw = sh_ms[w][0];
    const float e  = (mw == NEGINF) ? 0.f : __expf(mw - M);
    S += e * sh_ms[w][1];
    O += e * sh_O[w][tid];
  }
  outp[(b * H_ + h) * D_ + tid] = O / S;
}

extern "C" void kernel_launch(void* const* d_in, const int* in_sizes, int n_in,
                              void* d_out, int out_size, void* d_ws, size_t ws_size,
                              hipStream_t stream) {
  (void)n_in; (void)out_size; (void)d_ws; (void)ws_size;
  const float* q            = (const float*)d_in[0];
  const float* k_cache      = (const float*)d_in[1];
  const float* v_cache      = (const float*)d_in[2];
  const int*   block_tables = (const int*)d_in[3];
  const int*   context_lens = (const int*)d_in[4];
  const int*   crow         = (const int*)d_in[5];
  const int*   col          = (const int*)d_in[6];
  const int    col_stride   = in_sizes[6] / H_;

  dim3 grid(H_, B_);
  sparse_attn_decode<<<grid, 128, 0, stream>>>(
      q, k_cache, v_cache, block_tables, context_lens, crow, col, col_stride,
      (float*)d_out);
}
